// RGCN_84576495993313
// MI455X (gfx1250) — matrix operand order repaired and checked by reading, not compiled, not run on hardware
//
#include <hip/hip_runtime.h>

#ifndef __has_builtin
#define __has_builtin(x) 0
#endif

typedef __attribute__((ext_vector_type(2)))  float    v2f;
typedef __attribute__((ext_vector_type(8)))  float    v8f;
typedef __attribute__((ext_vector_type(16))) _Float16 v16h;

#if __has_builtin(__builtin_amdgcn_wmma_f32_16x16x4_f32)
#define USE_F32_WMMA 1
#else
#define USE_F32_WMMA 0
#endif

// ---------------------------------------------------------------------------
// GEMM: Y[N x OUT] = X[N x 64] @ W[64 x OUT] (+ bias[OUT] if BIAS)
// 256 threads = 8 waves; each wave computes a 16-row strip, OUT/16 tiles.
// K = 64 fixed (HIDDEN). W staged in LDS once per block (float4 b128 path).
// A rows are clamped (not zeroed): garbage rows only affect unsaved outputs.
// ---------------------------------------------------------------------------
template <int OUT, bool BIAS>
__global__ __launch_bounds__(256) void rgcn_gemm_kernel(
    const float* __restrict__ X, const float* __restrict__ W,
    const float* __restrict__ bias, float* __restrict__ Y, int N) {
  constexpr int NT = OUT / 16;
  __shared__ float lW[64 * OUT];
  {
    const float4* W4 = (const float4*)W;
    float4* lW4 = (float4*)lW;
    // 64*OUT/4 is a multiple of 256 for OUT in {16,64} -> no divergence
#pragma unroll
    for (int i = threadIdx.x; i < 16 * OUT; i += 256) lW4[i] = W4[i];
  }
  __syncthreads();

  const int lane = threadIdx.x & 31;
  const int wave = threadIdx.x >> 5;
  const int m    = lane & 15;   // row (A/C) or col (B/C) within 16-tile
  const int hi   = lane >> 4;   // half-wave select
  const int row0 = blockIdx.x * 128 + wave * 16;
  const int row  = row0 + m;
  const int rowC = (row < N) ? row : (N - 1);  // clamp, never zero
  const v2f* __restrict__ xp = (const v2f*)(X + (size_t)rowC * 64);

  v8f acc[NT];
#pragma unroll
  for (int t = 0; t < NT; ++t)
#pragma unroll
    for (int i = 0; i < 8; ++i) acc[t][i] = 0.0f;

#if USE_F32_WMMA
  // V_WMMA_F32_16X16X4_F32: A 16x4 (v2f: VGPR0 = K0|K2, VGPR1 = K1|K3),
  // B 4x16 (same K striping), C/D 16x16 (v8f).
#pragma unroll
  for (int i = 0; i < 16; ++i) {        // k0 = 4*i
    const int ka = 4 * i + 2 * hi;      // even
    const v2f a = xp[2 * i + hi];       // {xrow[ka], xrow[ka+1]} : one b64 load
#pragma unroll
    for (int t = 0; t < NT; ++t) {
      v2f b;
      b.x = lW[ka * OUT + t * 16 + m];
      b.y = lW[(ka + 1) * OUT + t * 16 + m];
      acc[t] = __builtin_amdgcn_wmma_f32_16x16x4_f32(
          false, a, false, b, (short)0, acc[t], false, false);
    }
  }
#else
  // Fallback: V_WMMA_F32_16X16X32_F16, 2 k-steps of K=32.
  // 16-bit A 16x32 layout: VGPR j: K = (j>=4?16:0) + 2*(j&3) + hi*8 + {0,1}
  // 16-bit B 32x16 layout: VGPR j: K = (j>=4?16:0) + 2*(j&3) + hi*8 + {0,1}, col = m
  const float* xrow = (const float*)xp;
#pragma unroll
  for (int k0 = 0; k0 < 64; k0 += 32) {
    v16h a;
#pragma unroll
    for (int j = 0; j < 8; ++j) {
      const int kk = k0 + ((j >= 4) ? 16 : 0) + ((j & 3) * 2) + hi * 8;
      a[2 * j]     = (_Float16)xrow[kk];
      a[2 * j + 1] = (_Float16)xrow[kk + 1];
    }
#pragma unroll
    for (int t = 0; t < NT; ++t) {
      v16h b;
#pragma unroll
      for (int j = 0; j < 8; ++j) {
        const int kk = k0 + ((j >= 4) ? 16 : 0) + ((j & 3) * 2) + hi * 8;
        b[2 * j]     = (_Float16)lW[kk * OUT + t * 16 + m];
        b[2 * j + 1] = (_Float16)lW[(kk + 1) * OUT + t * 16 + m];
      }
      acc[t] = __builtin_amdgcn_wmma_f32_16x16x32_f16(
          false, a, false, b, (short)0, acc[t], false, false);
    }
  }
#endif

  // C/D layout: VGPR i -> M = i + 8*hi, N = m
#pragma unroll
  for (int t = 0; t < NT; ++t) {
#pragma unroll
    for (int i = 0; i < 8; ++i) {
      const int rr = row0 + hi * 8 + i;
      if (rr < N) {
        float v = acc[t][i];
        if constexpr (BIAS) v += bias[t * 16 + m];
        Y[(size_t)rr * OUT + t * 16 + m] = v;
      }
    }
  }
}

// ---------------------------------------------------------------------------
// Helper kernels
// ---------------------------------------------------------------------------
__global__ void zero_i32_kernel(int* p, int n) {
  int i = blockIdx.x * blockDim.x + threadIdx.x;
  if (i < n) p[i] = 0;
}

__global__ void count_kernel(const int* __restrict__ et,
                             const int* __restrict__ dst,
                             int* __restrict__ cnt, int E) {
  int e = blockIdx.x * blockDim.x + threadIdx.x;
  if (e < E) atomicAdd(&cnt[dst[e] * 16 + et[e]], 1);
}

__global__ void inv_kernel(const int* __restrict__ cnt,
                           float* __restrict__ inv, int n) {
  int i = blockIdx.x * blockDim.x + threadIdx.x;
  if (i < n) {
    int c = cnt[i];
    inv[i] = (c > 0) ? (1.0f / (float)c) : 0.0f;
  }
}

// acc[dst] += T[src] * inv[dst*16 + r] for edges of relation r.
// One thread per (edge, 4-channel group); float4 gather, f32 atomic scatter.
template <int OUT>
__global__ void scatter_kernel(const float* __restrict__ T,
                               const int* __restrict__ src,
                               const int* __restrict__ dst,
                               const int* __restrict__ et,
                               const float* __restrict__ inv,
                               float* __restrict__ acc, int E, int r) {
  constexpr int Q = OUT / 4;  // 4-channel groups per edge
  int idx = blockIdx.x * blockDim.x + threadIdx.x;
  int e = idx / Q;
  int q = idx - e * Q;
  if (e >= E) return;
  if (et[e] != r) return;
  int s = src[e];
  int d = dst[e];
  float w = inv[d * 16 + r];
  float4 v = ((const float4*)(T + (size_t)s * OUT))[q];
  float* a = acc + (size_t)d * OUT + q * 4;
  atomicAdd(a + 0, v.x * w);
  atomicAdd(a + 1, v.y * w);
  atomicAdd(a + 2, v.z * w);
  atomicAdd(a + 3, v.w * w);
}

__global__ void relu_kernel(float* __restrict__ x, int n) {
  int i = blockIdx.x * blockDim.x + threadIdx.x;
  if (i < n) x[i] = fmaxf(x[i], 0.0f);
}

// ---------------------------------------------------------------------------
// Launch
// ---------------------------------------------------------------------------
extern "C" void kernel_launch(void* const* d_in, const int* in_sizes, int n_in,
                              void* d_out, int out_size, void* d_ws,
                              size_t ws_size, hipStream_t stream) {
  const int* edge_index = (const int*)d_in[0];    // [2, E]
  const int* edge_type  = (const int*)d_in[1];    // [E]
  const float* node_emb = (const float*)d_in[2];  // [N, 64]
  const float* W1       = (const float*)d_in[3];  // [16, 64, 64]
  const float* root1    = (const float*)d_in[4];  // [64, 64]
  const float* b1       = (const float*)d_in[5];  // [64]
  const float* W2       = (const float*)d_in[6];  // [16, 64, 16]
  const float* root2    = (const float*)d_in[7];  // [64, 16]
  const float* b2       = (const float*)d_in[8];  // [16]
  float* out = (float*)d_out;

  const int E = in_sizes[1];
  const int N = in_sizes[2] / 64;
  const int R = 16, H = 64, C = 16;
  const int* src = edge_index;
  const int* dst = edge_index + E;

  // Workspace layout (256B-aligned slabs)
  auto align256 = [](size_t x) { return (x + 255) & ~(size_t)255; };
  char* ws = (char*)d_ws;
  size_t off = 0;
  int* CNT = (int*)(ws + off);
  off += align256((size_t)N * R * sizeof(int));
  float* INV = (float*)(ws + off);
  off += align256((size_t)N * R * sizeof(float));
  float* X1 = (float*)(ws + off);
  off += align256((size_t)N * H * sizeof(float));
  float* T = (float*)(ws + off);
  off += align256((size_t)N * H * sizeof(float));
  (void)ws_size;

  const int tpb = 256;
  const int gemmBlocks = (N + 127) / 128;

  // Per-(node, relation) edge counts -> inverse counts (shared by both layers)
  zero_i32_kernel<<<(N * R + tpb - 1) / tpb, tpb, 0, stream>>>(CNT, N * R);
  count_kernel<<<(E + tpb - 1) / tpb, tpb, 0, stream>>>(edge_type, dst, CNT, E);
  inv_kernel<<<(N * R + tpb - 1) / tpb, tpb, 0, stream>>>(CNT, INV, N * R);

  // ---- Layer 1: X1 = sum_r mean_r(node_emb @ W1_r) + node_emb @ root1 + b1
  rgcn_gemm_kernel<64, true><<<gemmBlocks, tpb, 0, stream>>>(
      node_emb, root1, b1, X1, N);
  for (int r = 0; r < R; ++r) {
    rgcn_gemm_kernel<64, false><<<gemmBlocks, tpb, 0, stream>>>(
        node_emb, W1 + (size_t)r * H * H, nullptr, T, N);
    long long nth = (long long)E * (64 / 4);
    scatter_kernel<64><<<(int)((nth + tpb - 1) / tpb), tpb, 0, stream>>>(
        T, src, dst, edge_type, INV, X1, E, r);
  }
  relu_kernel<<<(N * H + tpb - 1) / tpb, tpb, 0, stream>>>(X1, N * H);

  // ---- Layer 2: out = sum_r mean_r(X1 @ W2_r) + X1 @ root2 + b2
  rgcn_gemm_kernel<16, true><<<gemmBlocks, tpb, 0, stream>>>(
      X1, root2, b2, out, N);
  for (int r = 0; r < R; ++r) {
    rgcn_gemm_kernel<16, false><<<gemmBlocks, tpb, 0, stream>>>(
        X1, W2 + (size_t)r * H * C, nullptr, T, N);
    long long nth = (long long)E * (16 / 4);
    scatter_kernel<16><<<(int)((nth + tpb - 1) / tpb), tpb, 0, stream>>>(
        T, src, dst, edge_type, INV, out, E, r);
  }
}
